// Decoder_62079457296567
// MI455X (gfx1250) — compile-verified, matrix operand order checked by reading, not run on hardware
//
#include <hip/hip_runtime.h>
#include <hip/hip_bf16.h>

// ---------------------------------------------------------------------------
// Tacotron2-style decoder for MI455X (gfx1250, wave32, WMMA).
//
// v2: register-blocked WMMA GEMM (4 M-tiles/wave, B-fragment reuse, K-unroll 2
// -> 4 independent v_wmma chains), single-GEMM LSTM gates via persistent
// concatenated activation buffers (no accumulate epilogue, no branchy stores),
// prenet / mproj / output head hoisted off the 500-step sequential chain.
// ---------------------------------------------------------------------------

typedef __bf16 bf16;
typedef __attribute__((ext_vector_type(8)))  __bf16 v8bf;
typedef __attribute__((ext_vector_type(16))) __bf16 v16bf;
typedef __attribute__((ext_vector_type(8)))  float  v8f;

#define BB    64
#define TENC  300
#define TDEC  500
#define DD    512
#define MELD  80
#define PRE   256
#define KA    1280   // att-LSTM input width: PRE + DD + DD... (256 ctx? no: 256+512+512)
#define KD    1536   // dec-LSTM input width: DD + DD + DD

__device__ __forceinline__ v16bf pack16(v8bf lo, v8bf hi) {
  return __builtin_shufflevector(lo, hi, 0,1,2,3,4,5,6,7,8,9,10,11,12,13,14,15);
}

__device__ __forceinline__ float sigmoidf_(float x) {
  return 1.0f / (1.0f + __expf(-x));
}

// ---------------------------------------------------------------------------
// Register-blocked bf16 GEMM:  C[M,N] = A[M,K] * W[N,K]^T + bias  [relu]
//   A row-major (row stride lda), W row-major [N,K] (row stride ldw) — the
//   native torch [out,in] layout gives contiguous-K B-fragments, no transpose.
// Block = 256 thr = 8 waves; wave owns 64(M) x 16(N): 4 accumulators, one
// shared B fragment per K-step -> 4-way WMMA ILP + 4x B reuse.
// Requirements: M % 64 == 0, K % 32 == 0 (caller pads); N edge via early-out.
// ---------------------------------------------------------------------------
__global__ __launch_bounds__(256) void gemm_bf16_kernel(
    const bf16* __restrict__ A, long lda,
    const bf16* __restrict__ W, long ldw,
    float* __restrict__ C, long ldc,
    const float* __restrict__ bias,
    bf16* __restrict__ Cbf,
    long N, long K, int relu)
{
  const int wave = threadIdx.x >> 5;
  const int lane = threadIdx.x & 31;
  const long n0 = (long)blockIdx.x * 128 + (long)wave * 16;
  const long m0 = (long)blockIdx.y * 64;
  if (n0 >= N) return;

  const int lm = lane & 15;        // M-row (A frag) / N-row (B frag) in tile
  const int g  = (lane >> 4) & 1;  // K-half selector

  const bf16* Ar0 = A + (m0 + lm) * lda + 8 * g;
  const bf16* Wr  = W + (n0 + lm) * ldw + 8 * g;
  const long  mstep = 16 * lda;

  v8f acc0 = {0.f,0.f,0.f,0.f,0.f,0.f,0.f,0.f};
  v8f acc1 = acc0, acc2 = acc0, acc3 = acc0;

#pragma unroll 2
  for (long k = 0; k < K; k += 32) {
    __builtin_prefetch(Wr + k + 64, 0, 1);          // global_prefetch_b8
    v16bf bf = pack16(*(const v8bf*)(Wr + k),
                      *(const v8bf*)(Wr + k + 16));
    v16bf a0 = pack16(*(const v8bf*)(Ar0 + k),
                      *(const v8bf*)(Ar0 + k + 16));
    v16bf a1 = pack16(*(const v8bf*)(Ar0 + mstep + k),
                      *(const v8bf*)(Ar0 + mstep + k + 16));
    v16bf a2 = pack16(*(const v8bf*)(Ar0 + 2 * mstep + k),
                      *(const v8bf*)(Ar0 + 2 * mstep + k + 16));
    v16bf a3 = pack16(*(const v8bf*)(Ar0 + 3 * mstep + k),
                      *(const v8bf*)(Ar0 + 3 * mstep + k + 16));
    acc0 = __builtin_amdgcn_wmma_f32_16x16x32_bf16(false, a0, false, bf, (short)0, acc0, false, false);
    acc1 = __builtin_amdgcn_wmma_f32_16x16x32_bf16(false, a1, false, bf, (short)0, acc1, false, false);
    acc2 = __builtin_amdgcn_wmma_f32_16x16x32_bf16(false, a2, false, bf, (short)0, acc2, false, false);
    acc3 = __builtin_amdgcn_wmma_f32_16x16x32_bf16(false, a3, false, bf, (short)0, acc3, false, false);
  }

  const long col = n0 + lm;
  const float bv = bias[col];
  v8f accs[4] = {acc0, acc1, acc2, acc3};
#pragma unroll
  for (int mt = 0; mt < 4; ++mt) {
#pragma unroll
    for (int r = 0; r < 8; ++r) {
      long row = m0 + mt * 16 + r + 8 * g;
      long off = row * ldc + col;
      float v = accs[mt][r] + bv;
      if (relu) v = v > 0.f ? v : 0.f;
      C[off] = v;
      if (Cbf) Cbf[off] = (bf16)v;
    }
  }
}

// ---------------------------------------------------------------------------
// LSTM pointwise: gates[B,2048] (i|f|g|o) -> c (fp32) + bf16 h written into up
// to two destination slots (concat-buffer slot and/or per-step archive).
// ---------------------------------------------------------------------------
__global__ void lstm_pointwise_kernel(
    const float* __restrict__ gates, float* __restrict__ c,
    bf16* __restrict__ h0, long h0s,
    bf16* __restrict__ h1, long h1s)
{
  int idx = blockIdx.x * blockDim.x + threadIdx.x;
  if (idx >= BB * DD) return;
  int b = idx >> 9, d = idx & (DD - 1);
  const float* gb = gates + (long)b * 2048;
  float gi = gb[d];
  float gf = gb[d + 512];
  float gg = gb[d + 1024];
  float go = gb[d + 1536];
  float cn = sigmoidf_(gf) * c[idx] + sigmoidf_(gi) * tanhf(gg);
  float hn = sigmoidf_(go) * tanhf(cn);
  c[idx] = cn;
  bf16 hb = (bf16)hn;
  h0[(long)b * h0s + d] = hb;
  if (h1) h1[(long)b * h1s + d] = hb;
}

// ---------------------------------------------------------------------------
// Fused location-sensitive attention (one WG of 256 thr / batch row):
//  conv(a,a_cum) -> LDS bf16 -> WMMA x loc_proj_w (K=32 exact, unroll 2) fused
//  with tanh(q+mproj+loc).v -> ds_add_f32 energies -> softmax -> a, a_cum,
//  ctx written as bf16 into both concat activation buffers.
// ---------------------------------------------------------------------------
__global__ __launch_bounds__(256) void attention_kernel(
    float* __restrict__ a_g, float* __restrict__ acum_g,
    const float* __restrict__ q, const float* __restrict__ mproj,
    const float* __restrict__ enc,
    const float* __restrict__ convw, const float* __restrict__ convb,
    const bf16* __restrict__ projw, const float* __restrict__ projb,
    const float* __restrict__ vw,
    bf16* __restrict__ ctx0, long ctx0s,
    bf16* __restrict__ ctx1, long ctx1s)
{
  __shared__ float loc0[336];
  __shared__ float loc1[336];
  __shared__ __align__(16) bf16 convT[304 * 32];
  __shared__ float e[304];
  __shared__ float red[256];

  const int b = blockIdx.x;
  const int tid = threadIdx.x;

  for (int i = tid; i < 336; i += 256) {
    int ts = i - 15;
    bool ok = (ts >= 0 && ts < TENC);
    loc0[i] = ok ? a_g[b * TENC + ts] : 0.f;
    loc1[i] = ok ? acum_g[b * TENC + ts] : 0.f;
  }
  for (int i = tid; i < 304; i += 256) e[i] = 0.f;
  __syncthreads();

  // location conv (32 ch, 31 taps, 2 in-ch) -> convT[t*32+c] bf16
  for (int idx = tid; idx < 304 * 32; idx += 256) {
    int t = idx >> 5, c = idx & 31;
    float s = 0.f;
    if (t < TENC) {
      s = convb[c];
      const float* w0 = convw + c * 62;
      const float* w1 = w0 + 31;
#pragma unroll
      for (int k = 0; k < 31; ++k)
        s += w0[k] * loc0[t + k] + w1[k] * loc1[t + k];
    }
    convT[idx] = (bf16)s;
  }
  __syncthreads();

  const int wave = tid >> 5;
  const int lane = tid & 31;
  const int lm = lane & 15;
  const int g = (lane >> 4) & 1;

  for (int tt = wave; tt < 19; tt += 8) {
    int t0 = tt * 16;
    const bf16* ap = &convT[(t0 + lm) * 32 + 8 * g];
    v16bf af = pack16(*(const v8bf*)ap, *(const v8bf*)(ap + 16)); // ds_load_b128

    float esum[8] = {0.f,0.f,0.f,0.f,0.f,0.f,0.f,0.f};
#pragma unroll 2
    for (int n0 = 0; n0 < DD; n0 += 16) {
      const bf16* wp = projw + (n0 + lm) * 32 + 8 * g;
      v16bf bf = pack16(*(const v8bf*)wp, *(const v8bf*)(wp + 16));
      v8f acc = {0.f,0.f,0.f,0.f,0.f,0.f,0.f,0.f};
      acc = __builtin_amdgcn_wmma_f32_16x16x32_bf16(
          false, af, false, bf, (short)0, acc, false, false);
      int col = n0 + lm;
      float qv = q[b * DD + col] + projb[col];
      float vv = vw[col];
#pragma unroll
      for (int r = 0; r < 8; ++r) {
        int t = t0 + r + 8 * g;
        if (t < TENC) {
          float x = acc[r] + qv + mproj[((long)b * TENC + t) * DD + col];
          esum[r] += tanhf(x) * vv;
        }
      }
    }
#pragma unroll
    for (int r = 0; r < 8; ++r) {
      int t = t0 + r + 8 * g;
      if (t < TENC) atomicAdd(&e[t], esum[r]);    // ds_add_f32
    }
  }
  __syncthreads();

  // softmax over e[0..299]
  float mx = -1e30f;
  for (int i = tid; i < TENC; i += 256) mx = fmaxf(mx, e[i]);
  red[tid] = mx; __syncthreads();
  for (int s = 128; s > 0; s >>= 1) {
    if (tid < s) red[tid] = fmaxf(red[tid], red[tid + s]);
    __syncthreads();
  }
  mx = red[0]; __syncthreads();

  float sm = 0.f;
  for (int i = tid; i < TENC; i += 256) {
    float ex = __expf(e[i] - mx);
    e[i] = ex;
    sm += ex;
  }
  red[tid] = sm; __syncthreads();
  for (int s = 128; s > 0; s >>= 1) {
    if (tid < s) red[tid] += red[tid + s];
    __syncthreads();
  }
  float inv = 1.f / red[0]; __syncthreads();

  for (int i = tid; i < TENC; i += 256) {
    float an = e[i] * inv;
    e[i] = an;
    a_g[b * TENC + i] = an;
    acum_g[b * TENC + i] += an;
  }
  __syncthreads();

  // context -> bf16 into both concat activation buffers
  for (int d = tid; d < DD; d += 256) {
    float s = 0.f;
    const float* ep = enc + (long)b * TENC * DD + d;
    for (int t = 0; t < TENC; ++t) s += e[t] * ep[(long)t * DD];
    bf16 sb = (bf16)s;
    ctx0[(long)b * ctx0s + d] = sb;
    ctx1[(long)b * ctx1s + d] = sb;
  }
}

// ---------------------------------------------------------------------------
// utility kernels
// ---------------------------------------------------------------------------

// fp32 -> bf16 into a strided destination (column offset + row stride)
__global__ void cvt_into_kernel(const float* __restrict__ src, bf16* __restrict__ dst,
                                long rows, long scols, long dld, long coff)
{
  long idx = (long)blockIdx.x * blockDim.x + threadIdx.x;
  if (idx >= rows * scols) return;
  long r = idx / scols, c = idx - r * scols;
  dst[r * dld + coff + c] = (bf16)src[idx];
}

__global__ void zero_f32_kernel(float* p, long n) {
  long i = (long)blockIdx.x * blockDim.x + threadIdx.x;
  if (i < n) p[i] = 0.f;
}

__global__ void zero_bf_kernel(bf16* p, long n) {
  long i = (long)blockIdx.x * blockDim.x + threadIdx.x;
  if (i < n) p[i] = (bf16)0.f;
}

__global__ void vadd_kernel(const float* a, const float* b, float* o, long n) {
  long i = (long)blockIdx.x * blockDim.x + threadIdx.x;
  if (i < n) o[i] = a[i] + b[i];
}

__global__ void build_outb_kernel(const float* mel_b, const float* stop_b, float* o) {
  int j = threadIdx.x;
  if (j < 96) o[j] = (j < MELD) ? mel_b[j] : ((j == MELD) ? stop_b[0] : 0.f);
}

// copy this step's prenet slice into the att concat buffer (cols 0..255)
__global__ void copy_pm_kernel(const bf16* __restrict__ src, bf16* __restrict__ dst) {
  int idx = blockIdx.x * blockDim.x + threadIdx.x;   // 64*256
  int b = idx >> 8, j = idx & 255;
  dst[(long)b * KA + j] = src[(long)b * TDEC * PRE + j];
}

// scatter [32000,96] head output -> mel_out [64,500,80] ++ stop_out [64,500]
__global__ void scatter_out_kernel(const float* __restrict__ Cs, float* __restrict__ out) {
  long idx = (long)blockIdx.x * blockDim.x + threadIdx.x;
  if (idx >= (long)BB * TDEC * 81) return;
  long r = idx / 81;
  int j = (int)(idx - r * 81);
  long b = r / TDEC, t = r - b * TDEC;
  float v = Cs[r * 96 + j];
  if (j < MELD) out[(b * TDEC + t) * MELD + j] = v;
  else          out[(long)BB * TDEC * MELD + b * TDEC + t] = v;
}

// ---------------------------------------------------------------------------
// host orchestration
// ---------------------------------------------------------------------------
extern "C" void kernel_launch(void* const* d_in, const int* in_sizes, int n_in,
                              void* d_out, int out_size, void* d_ws, size_t ws_size,
                              hipStream_t stream) {
  (void)in_sizes; (void)n_in; (void)out_size; (void)ws_size;

  const float* enc       = (const float*)d_in[0];
  const float* mel_in    = (const float*)d_in[1];
  const float* prenet_w1 = (const float*)d_in[2];
  const float* prenet_b1 = (const float*)d_in[3];
  const float* prenet_w2 = (const float*)d_in[4];
  const float* prenet_b2 = (const float*)d_in[5];
  const float* att_wih   = (const float*)d_in[6];
  const float* att_whh   = (const float*)d_in[7];
  const float* att_bih   = (const float*)d_in[8];
  const float* att_bhh   = (const float*)d_in[9];
  const float* dec_wih   = (const float*)d_in[10];
  const float* dec_whh   = (const float*)d_in[11];
  const float* dec_bih   = (const float*)d_in[12];
  const float* dec_bhh   = (const float*)d_in[13];
  const float* q_w       = (const float*)d_in[14];
  const float* q_b       = (const float*)d_in[15];
  const float* mem_w     = (const float*)d_in[16];
  const float* mem_b     = (const float*)d_in[17];
  const float* loc_conv_w= (const float*)d_in[18];
  const float* loc_conv_b= (const float*)d_in[19];
  const float* loc_proj_w= (const float*)d_in[20];
  const float* loc_proj_b= (const float*)d_in[21];
  const float* v_w       = (const float*)d_in[22];
  /* v_b (d_in[23]) is softmax-invariant */
  const float* mel_b     = (const float*)d_in[25];
  const float* mel_w     = (const float*)d_in[24];
  const float* stop_w    = (const float*)d_in[26];
  const float* stop_b    = (const float*)d_in[27];
  float* out = (float*)d_out;

  char* base = (char*)d_ws;
  size_t off = 0;
  auto alloc = [&](size_t bytes) -> char* {
    off = (off + 255) & ~(size_t)255;
    char* p = base + off;
    off += bytes;
    return p;
  };
  const long MROWS = (long)BB * TDEC;       // 32000
  const long EROWS = (long)BB * TENC;       // 19200

  bf16*  enc_bf    = (bf16*)alloc(EROWS * DD * 2);
  bf16*  melpad    = (bf16*)alloc(MROWS * 96 * 2);
  bf16*  pre1_bf   = (bf16*)alloc(MROWS * PRE * 2);
  bf16*  pm_all    = (bf16*)alloc(MROWS * PRE * 2);
  float* mproj     = (float*)alloc(EROWS * DD * 4);
  bf16*  h_dec_all = (bf16*)alloc(MROWS * DD * 2);
  float* Cscr      = (float*)alloc(MROWS * 256 * 4);
  bf16*  w_att_cat = (bf16*)alloc(2048L * KA * 2);   // [att_wih | att_whh]
  bf16*  w_dec_cat = (bf16*)alloc(2048L * KD * 2);   // [dec_wih | dec_whh]
  bf16*  w_q       = (bf16*)alloc(512L * 512 * 2);
  bf16*  w_mem     = (bf16*)alloc(512L * 512 * 2);
  bf16*  w_p1      = (bf16*)alloc(256L * 96 * 2);
  bf16*  w_p2      = (bf16*)alloc(256L * 256 * 2);
  bf16*  w_proj    = (bf16*)alloc(512L * 32 * 2);
  bf16*  w_out     = (bf16*)alloc(96L * 512 * 2);
  float* b_att     = (float*)alloc(2048 * 4);
  float* b_dec     = (float*)alloc(2048 * 4);
  float* b_out     = (float*)alloc(96 * 4);
  float* gates     = (float*)alloc((long)BB * 2048 * 4);
  float* c_att     = (float*)alloc((long)BB * DD * 4);
  float* c_dec     = (float*)alloc((long)BB * DD * 4);
  float* qbuf      = (float*)alloc((long)BB * DD * 4);
  bf16*  xa_cat    = (bf16*)alloc((long)BB * KA * 2); // [pm | ctx | h_att]
  bf16*  xd_cat    = (bf16*)alloc((long)BB * KD * 2); // [h_att | ctx | h_dec]
  float* a_buf     = (float*)alloc((long)BB * TENC * 4);
  float* acum_buf  = (float*)alloc((long)BB * TENC * 4);

  auto gemm = [&](const bf16* A, long lda, const bf16* W, long ldw,
                  float* C, long ldc, const float* bias, bf16* Cbf,
                  long M, long N, long K, int relu) {
    dim3 grid((unsigned)((N + 127) / 128), (unsigned)(M / 64));
    gemm_bf16_kernel<<<grid, 256, 0, stream>>>(A, lda, W, ldw, C, ldc,
                                               bias, Cbf, N, K, relu);
  };
  auto cvt = [&](const float* s, bf16* d, long r, long sc, long dld, long coff) {
    long n = r * sc;
    cvt_into_kernel<<<(unsigned)((n + 255) / 256), 256, 0, stream>>>(s, d, r, sc, dld, coff);
  };
  auto z32 = [&](float* p, long n) {
    zero_f32_kernel<<<(unsigned)((n + 255) / 256), 256, 0, stream>>>(p, n);
  };
  auto zbf = [&](bf16* p, long n) {
    zero_bf_kernel<<<(unsigned)((n + 255) / 256), 256, 0, stream>>>(p, n);
  };

  // --- setup: bf16 conversions, weight concat, padded buffers, zero state ---
  cvt(enc, enc_bf, EROWS, DD, DD, 0);
  zbf(melpad, MROWS * 96);
  cvt(mel_in, melpad, MROWS, MELD, 96, 0);
  cvt(att_wih, w_att_cat, 2048, 768, KA, 0);
  cvt(att_whh, w_att_cat, 2048, 512, KA, 768);
  cvt(dec_wih, w_dec_cat, 2048, 1024, KD, 0);
  cvt(dec_whh, w_dec_cat, 2048, 512, KD, 1024);
  cvt(q_w, w_q, 512, 512, 512, 0);
  cvt(mem_w, w_mem, 512, 512, 512, 0);
  zbf(w_p1, 256L * 96);
  cvt(prenet_w1, w_p1, 256, MELD, 96, 0);
  cvt(prenet_w2, w_p2, 256, 256, 256, 0);
  cvt(loc_proj_w, w_proj, 512, 32, 32, 0);
  zbf(w_out, 96L * 512);
  cvt(mel_w, w_out, MELD, 512, 512, 0);
  cvt(stop_w, w_out + (long)MELD * 512, 1, 512, 512, 0);

  vadd_kernel<<<8, 256, 0, stream>>>(att_bih, att_bhh, b_att, 2048);
  vadd_kernel<<<8, 256, 0, stream>>>(dec_bih, dec_bhh, b_dec, 2048);
  build_outb_kernel<<<1, 96, 0, stream>>>(mel_b, stop_b, b_out);

  z32(c_att, (long)BB * DD);
  z32(c_dec, (long)BB * DD);
  z32(a_buf, (long)BB * TENC);
  z32(acum_buf, (long)BB * TENC);
  zbf(xa_cat, (long)BB * KA);
  zbf(xd_cat, (long)BB * KD);

  // --- hoisted GEMMs (off the sequential chain) ---
  gemm(enc_bf, DD, w_mem, DD, mproj, DD, mem_b, nullptr, EROWS, DD, DD, 0);
  gemm(melpad, 96, w_p1, 96, Cscr, PRE, prenet_b1, pre1_bf, MROWS, PRE, 96, 1);
  gemm(pre1_bf, PRE, w_p2, PRE, Cscr, PRE, prenet_b2, pm_all, MROWS, PRE, PRE, 1);

  // --- sequential decode loop (7 launches / step) ---
  for (int t = 0; t < TDEC; ++t) {
    // xa_cat[:, 0:256] = pm_all[:, t, :]
    copy_pm_kernel<<<BB, 256, 0, stream>>>(pm_all + (long)t * PRE, xa_cat);

    // attention-LSTM gates: xa_cat[64,1280] @ w_att_cat^T + (bih+bhh)
    gemm(xa_cat, KA, w_att_cat, KA, gates, 2048, b_att, nullptr,
         BB, 2048, KA, 0);
    // h_att -> xa_cat slot (recurrent, next step) + xd_cat slot (this step)
    lstm_pointwise_kernel<<<BB, DD, 0, stream>>>(
        gates, c_att, xa_cat + 768, KA, xd_cat, KD);

    // query: q = h_att @ q_w^T + q_b  (A = xd_cat cols 0..511)
    gemm(xd_cat, KD, w_q, DD, qbuf, DD, q_b, nullptr, BB, DD, DD, 0);

    // fused attention -> a, a_cum, ctx (into both concat buffers)
    attention_kernel<<<BB, 256, 0, stream>>>(
        a_buf, acum_buf, qbuf, mproj, enc, loc_conv_w, loc_conv_b,
        w_proj, loc_proj_b, v_w, xa_cat + 256, KA, xd_cat + 512, KD);

    // decoder-LSTM gates: xd_cat[64,1536] @ w_dec_cat^T + (bih+bhh)
    gemm(xd_cat, KD, w_dec_cat, KD, gates, 2048, b_dec, nullptr,
         BB, 2048, KD, 0);
    // h_dec -> xd_cat slot (recurrent) + per-step archive for output head
    lstm_pointwise_kernel<<<BB, DD, 0, stream>>>(
        gates, c_dec, xd_cat + 1024, KD, h_dec_all + (long)t * DD, (long)TDEC * DD);
  }

  // --- deferred output head over all saved h_dec states ---
  gemm(h_dec_all, DD, w_out, DD, Cscr, 96, b_out, nullptr, MROWS, 96, DD, 0);
  scatter_out_kernel<<<(unsigned)(((long)BB * TDEC * 81 + 255) / 256), 256, 0,
                       stream>>>(Cscr, out);
}
